// Decoder_40578851012890
// MI455X (gfx1250) — matrix operand order repaired and checked by reading, not compiled, not run on hardware
//
#include <hip/hip_runtime.h>
#include <hip/hip_bf16.h>

#define B_    1024
#define NG_   32
#define G_    32
#define E_    64
#define H_    128
#define PP_   512
#define BOT_  1024
#define MLPD_ 1024
#define SEQ_  12

typedef __attribute__((ext_vector_type(16))) __bf16 v16bf;
typedef __attribute__((ext_vector_type(8)))  float  v8f;

union BF16Frag { v16bf v; unsigned short u[16]; unsigned int w[8]; };

__device__ __forceinline__ unsigned short f2bf(float f) {
    unsigned u = __float_as_uint(f);
    unsigned r = (u >> 16) & 1u;
    u += 0x7fffu + r;                 // round-to-nearest-even
    return (unsigned short)(u >> 16);
}

// ---------------------------------------------------------------------------
// One-time weight prep: transpose to [N,K] bf16 so each B-fragment VGPR is one
// contiguous dword load; fold W_sp / b_sp / b_pp1 into a rank-2 update + bias.
// ---------------------------------------------------------------------------
__global__ void prep_weights(const float* __restrict__ W_m1, const float* __restrict__ W_m2,
                             const float* __restrict__ W_pp2, const float* __restrict__ W_pp1,
                             const float* __restrict__ W_sp, const float* __restrict__ b_sp,
                             const float* __restrict__ b_pp1,
                             unsigned short* __restrict__ Wm1T, unsigned short* __restrict__ Wm2T,
                             unsigned short* __restrict__ W2T,  unsigned short* __restrict__ W1bT,
                             float* __restrict__ Wsp1, float* __restrict__ bias1p)
{
    int stride = gridDim.x * blockDim.x;
    int t0 = blockIdx.x * blockDim.x + threadIdx.x;
    for (int idx = t0; idx < MLPD_ * (H_ + BOT_); idx += stride) {      // Wm1T [1024 x 1152]
        int n = idx / (H_ + BOT_), k = idx % (H_ + BOT_);
        Wm1T[idx] = f2bf(W_m1[k * MLPD_ + n]);
    }
    for (int idx = t0; idx < H_ * MLPD_; idx += stride) {               // Wm2T [128 x 1024]
        int n = idx / MLPD_, k = idx % MLPD_;
        Wm2T[idx] = f2bf(W_m2[k * H_ + n]);
    }
    for (int idx = t0; idx < BOT_ * PP_; idx += stride) {               // W2T [1024 x 512]
        int n = idx / PP_, k = idx % PP_;
        W2T[idx] = f2bf(W_pp2[k * BOT_ + n]);
    }
    for (int idx = t0; idx < PP_ * H_; idx += stride) {                 // W1bT [512 x 128]
        int n = idx / H_, r = idx % H_;
        W1bT[idx] = f2bf(W_pp1[(E_ + r) * PP_ + n]);
    }
    for (int idx = t0; idx < 2 * PP_; idx += stride) {                  // Wsp1 = W_sp @ W_pp1[:64]
        int d = idx >> 9, c = idx & 511;
        float s = 0.f;
        for (int e = 0; e < E_; ++e) s += W_sp[d * E_ + e] * W_pp1[e * PP_ + c];
        Wsp1[idx] = s;
    }
    for (int idx = t0; idx < PP_; idx += stride) {                      // bias1p = b_sp@W_pp1[:64] + b_pp1
        float s = b_pp1[idx];
        for (int e = 0; e < E_; ++e) s += b_sp[e] * W_pp1[e * PP_ + idx];
        bias1p[idx] = s;
    }
}

// ---------------------------------------------------------------------------
// Initial state: din0 = last_pos_rel@W_emb+b_emb ; copy h0/c0/last_pos
// ---------------------------------------------------------------------------
__global__ void init_state(const float* __restrict__ last_pos, const float* __restrict__ last_pos_rel,
                           const float* __restrict__ h0, const float* __restrict__ c0,
                           const float* __restrict__ W_emb, const float* __restrict__ b_emb,
                           float* __restrict__ curr, float* __restrict__ cbuf,
                           float* __restrict__ din, float* __restrict__ hbuf)
{
    int b = blockIdx.x, tid = threadIdx.x;
    float r0 = last_pos_rel[b * 2 + 0], r1 = last_pos_rel[b * 2 + 1];
    if (tid < E_)
        din[b * E_ + tid] = r0 * W_emb[tid] + r1 * W_emb[E_ + tid] + b_emb[tid];
    if (tid < H_) { hbuf[b * H_ + tid] = h0[b * H_ + tid]; cbuf[b * H_ + tid] = c0[b * H_ + tid]; }
    if (tid < 2)  curr[b * 2 + tid] = last_pos[b * 2 + tid];
}

// ---------------------------------------------------------------------------
// LSTM cell + rel_pos head + next embedding (small: VALU, one block per row)
// writes bf16 h into hcat[:,0:128] for the WMMA consumers.
// ---------------------------------------------------------------------------
__global__ void lstm_step(const float* __restrict__ W_ih, const float* __restrict__ b_ih,
                          const float* __restrict__ W_hh, const float* __restrict__ b_hh,
                          const float* __restrict__ W_h2p, const float* __restrict__ b_h2p,
                          const float* __restrict__ W_emb, const float* __restrict__ b_emb,
                          const float* __restrict__ hbuf, float* __restrict__ cbuf,
                          float* __restrict__ din, float* __restrict__ curr,
                          unsigned short* __restrict__ hcat, float* __restrict__ pred_t)
{
    __shared__ float sx[E_], sh[H_], sg[4 * H_], sh2[H_], srp[2];
    int b = blockIdx.x, tid = threadIdx.x;
    if (tid < E_) sx[tid] = din[b * E_ + tid];
    if (tid < H_) sh[tid] = hbuf[b * H_ + tid];
    __syncthreads();
    #pragma unroll
    for (int q = 0; q < 4; ++q) {
        int k = tid * 4 + q;
        float acc = b_ih[k] + b_hh[k];
        for (int e = 0; e < E_; ++e) acc += sx[e] * W_ih[k * E_ + e];
        for (int r = 0; r < H_; ++r) acc += sh[r] * W_hh[k * H_ + r];
        sg[k] = acc;
    }
    __syncthreads();
    if (tid < H_) {
        float ig = sg[tid], fg = sg[H_ + tid], gg = sg[2 * H_ + tid], og = sg[3 * H_ + tid];
        float si = 1.f / (1.f + __expf(-ig));
        float sf = 1.f / (1.f + __expf(-fg));
        float so = 1.f / (1.f + __expf(-og));
        float c2 = sf * cbuf[b * H_ + tid] + si * tanhf(gg);
        float h2 = so * tanhf(c2);
        cbuf[b * H_ + tid] = c2;
        sh2[tid] = h2;
        hcat[b * (H_ + BOT_) + tid] = f2bf(h2);
    }
    __syncthreads();
    if (tid < 2) {
        float rp = b_h2p[tid];
        for (int r = 0; r < H_; ++r) rp += sh2[r] * W_h2p[r * 2 + tid];
        srp[tid] = rp;
        pred_t[b * 2 + tid] = rp;                       // rel_pos output
        curr[b * 2 + tid] += rp;                        // curr_pos for pool net
    }
    __syncthreads();
    if (tid < E_)
        din[b * E_ + tid] = srp[0] * W_emb[tid] + srp[1] * W_emb[E_ + tid] + b_emb[tid];
}

// ---------------------------------------------------------------------------
// Generic bf16 WMMA GEMM:  out = [relu](A[M,K] @ B[K,N] + bias)
// A bf16 row-major (lda), B transposed bf16 [N,K].
// Register-blocked: each wave owns a 64x64 output (4x4 WMMA tiles, 16 v8f
// accumulators) -> every A fragment reused 4x, ~32 FLOP/B from L2.
// __launch_bounds__(128,1) unlocks the full VGPR budget so accumulators stay
// resident (no scratch spills); B fragments are loaded one-at-a-time to keep
// the peak live set at 4A + 1B + 16 acc.
// M, N must be multiples of 64; K multiple of 32.
// ---------------------------------------------------------------------------
__global__ void __launch_bounds__(128, 1)
gemm_bf16_wmma(const unsigned short* __restrict__ A, int lda,
               const unsigned short* __restrict__ BT,
               const float* __restrict__ bias,
               float* __restrict__ outF, unsigned short* __restrict__ outBf,
               int ldo, int M, int N, int K, int relu)
{
    int lane = threadIdx.x & 31;
    int wv   = threadIdx.x >> 5;
    int nB   = N >> 6;
    int wt   = blockIdx.x * 4 + wv;
    if (wt >= (M >> 6) * nB) return;                    // wave-uniform
    int tm0 = (wt / nB) << 6, tn0 = (wt % nB) << 6;

    const unsigned short* Abase = A  + (size_t)(tm0 + (lane & 15)) * lda;
    const unsigned short* Bbase = BT + (size_t)(tn0 + (lane & 15)) * K;
    int kaBase = (lane < 16) ? 0 : 8;                   // A: hi lanes hold K+8 block
    int kbBase = (lane < 16) ? 0 : 8;                   // B: hi lanes hold K+8 block

    v8f acc[4][4] = {};
    for (int kt = 0; kt < K; kt += 32) {
        BF16Frag a[4];
        #pragma unroll
        for (int mi = 0; mi < 4; ++mi) {
            const unsigned short* Ar = Abase + (size_t)mi * 16 * lda + kt;
            __builtin_prefetch(Ar + 32, 0, 1);          // -> global_prefetch_b8
            #pragma unroll
            for (int v = 0; v < 8; ++v) {
                int ka = ((v & 4) << 2) + kaBase + ((v & 3) << 1);   // {0,16}+{0,8}+2*(v%4)
                a[mi].w[v] = *(const unsigned int*)(Ar + ka);
            }
        }
        #pragma unroll
        for (int ni = 0; ni < 4; ++ni) {
            const unsigned short* Bc = Bbase + (size_t)ni * 16 * K + kt;
            __builtin_prefetch(Bc + 32, 0, 1);
            BF16Frag b;
            #pragma unroll
            for (int v = 0; v < 8; ++v)
                b.w[v] = *(const unsigned int*)(Bc + ((v & 4) << 2) + kbBase + ((v & 3) << 1));
            #pragma unroll
            for (int mi = 0; mi < 4; ++mi)
                acc[mi][ni] = __builtin_amdgcn_wmma_f32_16x16x32_bf16(
                    false, a[mi].v, false, b.v, (short)0, acc[mi][ni], false, false);
        }
    }
    #pragma unroll
    for (int ni = 0; ni < 4; ++ni) {
        int col = tn0 + ni * 16 + (lane & 15);
        float bn = bias ? bias[col] : 0.f;
        #pragma unroll
        for (int mi = 0; mi < 4; ++mi) {
            int rbase = tm0 + mi * 16 + ((lane < 16) ? 0 : 8);
            #pragma unroll
            for (int k = 0; k < 8; ++k) {
                float v = acc[mi][ni][k] + bn;
                if (relu) v = v > 0.f ? v : 0.f;
                size_t o = (size_t)(rbase + k) * ldo + col;
                if (outF) outF[o] = v; else outBf[o] = f2bf(v);
            }
        }
    }
}

// ---------------------------------------------------------------------------
// X[g,i,j,:] = relu( HW[g,j,:] + rel@Wsp1 + bias1p )  -> bf16 [32768,512]
// ---------------------------------------------------------------------------
__global__ void pool_build_x(const float* __restrict__ HW, const float* __restrict__ Wsp1,
                             const float* __restrict__ bias1p, const float* __restrict__ pos,
                             unsigned short* __restrict__ X)
{
    int r = blockIdx.x;                       // (g,i,j) flat
    int g = r >> 10, rem = r & 1023, i = rem >> 5, j = rem & 31;
    int pj = g * G_ + j, pi = g * G_ + i;
    float r0 = pos[pj * 2 + 0] - pos[pi * 2 + 0];
    float r1 = pos[pj * 2 + 1] - pos[pi * 2 + 1];
    const float* hwrow = HW + (size_t)pj * PP_;
    for (int c = threadIdx.x; c < PP_; c += blockDim.x) {
        float v = hwrow[c] + r0 * Wsp1[c] + r1 * Wsp1[PP_ + c] + bias1p[c];
        X[(size_t)r * PP_ + c] = f2bf(v > 0.f ? v : 0.f);
    }
}

// ---------------------------------------------------------------------------
// Pool layer 2 fused with neighbor-max:  pool = max_j relu(X@W_pp2 + b2).
// 64x64 per wave; the 64-row block covers ALL 32 neighbors of exactly two
// pedestrians, so the max completes in-wave: reduce 8 acc rows per lane,
// merge the two mi-tiles of each pedestrian, fold lanes L/L+16 via shfl_xor,
// then plain-store bf16 straight into hcat[:,128:]. No atomics, no memset,
// no pool buffer, no pack pass.
// ---------------------------------------------------------------------------
__global__ void __launch_bounds__(128, 1)
pool_gemm_max(const unsigned short* __restrict__ X,
              const unsigned short* __restrict__ W2T,
              const float* __restrict__ b2,
              unsigned short* __restrict__ hcat)
{
    int lane = threadIdx.x & 31;
    int wv   = threadIdx.x >> 5;
    int wt   = blockIdx.x * 4 + wv;           // 512 m-blocks x 16 n-blocks
    int tm0  = (wt >> 4) << 6;
    int tn0  = (wt & 15) << 6;

    const unsigned short* Abase = X   + (size_t)(tm0 + (lane & 15)) * PP_;
    const unsigned short* Bbase = W2T + (size_t)(tn0 + (lane & 15)) * PP_;
    int kaBase = (lane < 16) ? 0 : 8;
    int kbBase = (lane < 16) ? 0 : 8;

    v8f acc[4][4] = {};
    for (int kt = 0; kt < PP_; kt += 32) {
        BF16Frag a[4];
        #pragma unroll
        for (int mi = 0; mi < 4; ++mi) {
            const unsigned short* Ar = Abase + (size_t)mi * 16 * PP_ + kt;
            __builtin_prefetch(Ar + 32, 0, 1);
            #pragma unroll
            for (int v = 0; v < 8; ++v) {
                int ka = ((v & 4) << 2) + kaBase + ((v & 3) << 1);
                a[mi].w[v] = *(const unsigned int*)(Ar + ka);
            }
        }
        #pragma unroll
        for (int ni = 0; ni < 4; ++ni) {
            const unsigned short* Bc = Bbase + (size_t)ni * 16 * PP_ + kt;
            __builtin_prefetch(Bc + 32, 0, 1);
            BF16Frag b;
            #pragma unroll
            for (int v = 0; v < 8; ++v)
                b.w[v] = *(const unsigned int*)(Bc + ((v & 4) << 2) + kbBase + ((v & 3) << 1));
            #pragma unroll
            for (int mi = 0; mi < 4; ++mi)
                acc[mi][ni] = __builtin_amdgcn_wmma_f32_16x16x32_bf16(
                    false, a[mi].v, false, b.v, (short)0, acc[mi][ni], false, false);
        }
    }
    int ped0 = tm0 >> 5;                      // two pedestrians per wave block
    #pragma unroll
    for (int ni = 0; ni < 4; ++ni) {
        int col = tn0 + ni * 16 + (lane & 15);
        float bn = b2[col];
        #pragma unroll
        for (int pp = 0; pp < 2; ++pp) {      // mi 0,1 -> ped0 ; mi 2,3 -> ped0+1
            float m = 0.f;                    // relu floor; max(0,max(v)) == max(relu(v))
            #pragma unroll
            for (int mi = pp * 2; mi < pp * 2 + 2; ++mi)
                #pragma unroll
                for (int k = 0; k < 8; ++k)
                    m = fmaxf(m, acc[mi][ni][k] + bn);
            m = fmaxf(m, __shfl_xor(m, 16, 32));      // fold row halves (wave32)
            if (lane < 16)
                hcat[(size_t)(ped0 + pp) * (H_ + BOT_) + H_ + col] = f2bf(m);
        }
    }
}

__global__ void copy_f32(const float* __restrict__ src, float* __restrict__ dst, int n)
{
    int i = blockIdx.x * blockDim.x + threadIdx.x;
    if (i < n) dst[i] = src[i];
}

// ---------------------------------------------------------------------------
extern "C" void kernel_launch(void* const* d_in, const int* in_sizes, int n_in,
                              void* d_out, int out_size, void* d_ws, size_t ws_size,
                              hipStream_t stream) {
    (void)in_sizes; (void)n_in; (void)out_size; (void)ws_size;
    const float* last_pos     = (const float*)d_in[0];
    const float* last_pos_rel = (const float*)d_in[1];
    const float* decoder_h    = (const float*)d_in[2];
    const float* decoder_c    = (const float*)d_in[3];
    /* d_in[4] seq_start_end: contiguous equal groups, unused */
    const float* W_emb = (const float*)d_in[5];  const float* b_emb = (const float*)d_in[6];
    const float* W_ih  = (const float*)d_in[7];  const float* b_ih  = (const float*)d_in[8];
    const float* W_hh  = (const float*)d_in[9];  const float* b_hh  = (const float*)d_in[10];
    const float* W_h2p = (const float*)d_in[11]; const float* b_h2p = (const float*)d_in[12];
    const float* W_sp  = (const float*)d_in[13]; const float* b_sp  = (const float*)d_in[14];
    const float* W_pp1 = (const float*)d_in[15]; const float* b_pp1 = (const float*)d_in[16];
    const float* W_pp2 = (const float*)d_in[17]; const float* b_pp2 = (const float*)d_in[18];
    const float* W_m1  = (const float*)d_in[19]; const float* b_m1  = (const float*)d_in[20];
    const float* W_m2  = (const float*)d_in[21]; const float* b_m2  = (const float*)d_in[22];
    float* out = (float*)d_out;

    char* p = (char*)d_ws;
    auto alloc = [&](size_t bytes) { void* r = (void*)p; p += (bytes + 255) & ~(size_t)255; return r; };

    unsigned short* hcat  = (unsigned short*)alloc((size_t)B_ * (H_ + BOT_) * 2);
    unsigned short* act1  = (unsigned short*)alloc((size_t)B_ * MLPD_ * 2);
    unsigned short* Xbf   = (unsigned short*)alloc((size_t)NG_ * G_ * G_ * PP_ * 2);
    float*          HW    = (float*)alloc((size_t)B_ * PP_ * 4);
    float*          hbuf  = (float*)alloc((size_t)B_ * H_ * 4);
    float*          cbuf  = (float*)alloc((size_t)B_ * H_ * 4);
    float*          din   = (float*)alloc((size_t)B_ * E_ * 4);
    float*          curr  = (float*)alloc((size_t)B_ * 2 * 4);
    unsigned short* Wm1T  = (unsigned short*)alloc((size_t)MLPD_ * (H_ + BOT_) * 2);
    unsigned short* Wm2T  = (unsigned short*)alloc((size_t)H_ * MLPD_ * 2);
    unsigned short* W2T   = (unsigned short*)alloc((size_t)BOT_ * PP_ * 2);
    unsigned short* W1bT  = (unsigned short*)alloc((size_t)PP_ * H_ * 2);
    float*          Wsp1  = (float*)alloc(2 * PP_ * 4);
    float*          bias1p= (float*)alloc(PP_ * 4);

    prep_weights<<<2048, 256, 0, stream>>>(W_m1, W_m2, W_pp2, W_pp1, W_sp, b_sp, b_pp1,
                                           Wm1T, Wm2T, W2T, W1bT, Wsp1, bias1p);
    init_state<<<B_, 128, 0, stream>>>(last_pos, last_pos_rel, decoder_h, decoder_c,
                                       W_emb, b_emb, curr, cbuf, din, hbuf);

    for (int t = 0; t < SEQ_; ++t) {
        lstm_step<<<B_, 128, 0, stream>>>(W_ih, b_ih, W_hh, b_hh, W_h2p, b_h2p, W_emb, b_emb,
                                          hbuf, cbuf, din, curr, hcat, out + (size_t)t * B_ * 2);
        // HW = h @ W_pp1[64:,:]   (M=1024,K=128,N=512) : 16x8 = 128 waves
        gemm_bf16_wmma<<<32, 128, 0, stream>>>(hcat, H_ + BOT_, W1bT, nullptr,
                                               HW, nullptr, PP_, B_, PP_, H_, 0);
        pool_build_x<<<NG_ * G_ * G_, 256, 0, stream>>>(HW, Wsp1, bias1p, curr, Xbf);
        // relu(X @ W_pp2 + b2) + max over neighbors (M=32768,K=512,N=1024) : 8192 waves
        pool_gemm_max<<<2048, 128, 0, stream>>>(Xbf, W2T, b_pp2, hcat);
        // act1 = relu(hcat @ W_m1 + b_m1)   (M=1024,K=1152,N=1024) : 256 waves
        gemm_bf16_wmma<<<64, 128, 0, stream>>>(hcat, H_ + BOT_, Wm1T, b_m1,
                                               nullptr, act1, MLPD_, B_, MLPD_, H_ + BOT_, 1);
        // h = relu(act1 @ W_m2 + b_m2)      (M=1024,K=1024,N=128) : 32 waves
        gemm_bf16_wmma<<<8, 128, 0, stream>>>(act1, MLPD_, Wm2T, b_m2,
                                              hbuf, nullptr, H_, B_, H_, MLPD_, 1);
    }
    copy_f32<<<(B_ * H_ + 255) / 256, 256, 0, stream>>>(hbuf, out + (size_t)SEQ_ * B_ * 2, B_ * H_);
}
